// HomoAttention_3023656976710
// MI455X (gfx1250) — matrix operand order repaired and checked by reading, not compiled, not run on hardware
//
#include <hip/hip_runtime.h>
#include <hip/hip_bf16.h>
#include <math.h>

typedef __attribute__((ext_vector_type(16))) __bf16 v16bf;
typedef __attribute__((ext_vector_type(8)))  float  v8f;

#define B_   2
#define N_   6
#define FD   128
#define DIM  128
#define H_   64
#define W_   176
#define P_   (H_*W_)          // 11264
#define IMG_W_ 704.0f
#define IMG_H_ 256.0f
#define BP_  (B_*P_)          // 22528

// ---------------- async global->LDS copy (CDNA5 ASYNCcnt path) ----------------
#if __has_builtin(__builtin_amdgcn_global_load_async_to_lds_b128)
#define HAS_ASYNC_LDS 1
typedef int vint4_t __attribute__((vector_size(16)));
typedef __attribute__((address_space(1))) vint4_t* gb128_t;
typedef __attribute__((address_space(3))) vint4_t* lb128_t;
#else
#define HAS_ASYNC_LDS 0
#endif

// copy 16 bytes global -> LDS (8 bf16 elements)
__device__ inline void cp16(const __bf16* g, __bf16* l) {
#if HAS_ASYNC_LDS
    __builtin_amdgcn_global_load_async_to_lds_b128((gb128_t)(void*)g, (lb128_t)(void*)l, 0, 0);
#else
    *(int4*)l = *(const int4*)g;
#endif
}
__device__ inline void cp_wait() {
#if HAS_ASYNC_LDS
#if __has_builtin(__builtin_amdgcn_s_wait_asynccnt)
    __builtin_amdgcn_s_wait_asynccnt(0);
#else
    asm volatile("s_wait_asynccnt 0" ::: "memory");
#endif
#endif
}

// ---------------- WMMA fragment builders (16-bit, wave32) ----------------
// A matrix 16x32 (MxK): lane -> M = lane&15, K-half h = lane>>4.
// VGPR j (pair p): j<4 -> K = 2j+p+8h ; j>=4 -> K = 2j+p+8+8h.
__device__ inline v16bf frag_a(const __bf16* rowK, int h) {
    v16bf f;
#pragma unroll
    for (int j = 0; j < 8; ++j) {
        int k = (j < 4) ? (2*j + 8*h) : (2*j + 8 + 8*h);
        f[2*j]   = rowK[k];
        f[2*j+1] = rowK[k+1];
    }
    return f;
}
// B matrix 32x16 (KxN): lane -> N = lane&15, K-half h = lane>>4 (same K mapping as A).
__device__ inline v16bf frag_b(const __bf16* colK, int h) {
    v16bf f;
#pragma unroll
    for (int j = 0; j < 8; ++j) {
        int k = (j < 4) ? (2*j + 8*h) : (2*j + 8 + 8*h);
        f[2*j]   = colK[k];
        f[2*j+1] = colK[k+1];
    }
    return f;
}

// ---------------- K0: homographies + weight casts (+ pre-transpose MLP weights) ----------------
__global__ void setup_kernel(const float* __restrict__ I_src,
                             const float* __restrict__ I_tar_inv,
                             const float* __restrict__ E,
                             const float* __restrict__ dis,
                             const float* __restrict__ nrm,
                             const float* __restrict__ conv_w,
                             const float* __restrict__ w1,
                             const float* __restrict__ w2,
                             float* __restrict__ Hm,
                             __bf16* __restrict__ cwbf,
                             __bf16* __restrict__ w1T,   // [n=256][k=128]
                             __bf16* __restrict__ w2T) { // [n=128][k=256]
    int tid = threadIdx.x;
    if (tid < B_*N_) {
        int bb = tid / N_;
        const float* Ee  = E + tid*16;
        const float* Is  = I_src + tid*9;
        const float* Iti = I_tar_inv + bb*9;
        const float* nm  = nrm + bb*3;
        float dv = dis[bb];
        float M[9], tmp[9], Hh[9];
#pragma unroll
        for (int i = 0; i < 3; ++i)
#pragma unroll
            for (int j = 0; j < 3; ++j)
                M[i*3+j] = Ee[i*4+j] - Ee[i*4+3]*nm[j]/dv;
#pragma unroll
        for (int i = 0; i < 3; ++i)
#pragma unroll
            for (int j = 0; j < 3; ++j) {
                float s = 0.f;
#pragma unroll
                for (int k = 0; k < 3; ++k) s += Is[i*3+k]*M[k*3+j];
                tmp[i*3+j] = s;
            }
#pragma unroll
        for (int i = 0; i < 3; ++i)
#pragma unroll
            for (int j = 0; j < 3; ++j) {
                float s = 0.f;
#pragma unroll
                for (int k = 0; k < 3; ++k) s += tmp[i*3+k]*Iti[k*3+j];
                Hh[i*3+j] = s;
            }
        for (int k = 0; k < 9; ++k) Hm[tid*9+k] = Hh[k];
    }
    for (int i = tid; i < DIM*FD; i += blockDim.x) cwbf[i] = (__bf16)conv_w[i];
    // w1 (k=128 rows, n=256 cols) -> w1T[n*128 + k]
    for (int i = tid; i < 2*DIM*DIM; i += blockDim.x) {
        int k = i / (2*DIM), n = i % (2*DIM);
        w1T[(size_t)n*DIM + k] = (__bf16)w1[i];
    }
    // w2 (k=256 rows, n=128 cols) -> w2T[n*256 + k]
    for (int i = tid; i < 2*DIM*DIM; i += blockDim.x) {
        int k = i / DIM, n = i % DIM;
        w2T[(size_t)n*(2*DIM) + k] = (__bf16)w2[i];
    }
}

// ---------------- K1: BN + ReLU + cast to bf16, transposed to (bn, p, c) ----------------
__global__ void bnrelu_kernel(const float* __restrict__ feat,
                              const float* __restrict__ g, const float* __restrict__ be,
                              const float* __restrict__ mn, const float* __restrict__ vr,
                              __bf16* __restrict__ xfbf, int total) {
    for (int i = blockIdx.x*blockDim.x + threadIdx.x; i < total; i += gridDim.x*blockDim.x) {
        int bn = i / (FD*P_);
        int c  = (i / P_) % FD;
        int p  = i % P_;
        float x = (feat[i] - mn[c]) * rsqrtf(vr[c] + 1e-5f) * g[c] + be[c];
        xfbf[((size_t)bn*P_ + p)*FD + c] = (__bf16)fmaxf(x, 0.f);
    }
}

// ---------------- K2: 1x1 conv GEMM (WMMA bf16) -> val f32 ----------------
// Block: 128 out-channels x 64 pixels, K=128 in 4 WMMA steps. 8 waves, 16 rows each.
__global__ __launch_bounds__(256) void conv_kernel(const __bf16* __restrict__ xfbf,
                                                   const __bf16* __restrict__ cwbf,
                                                   float* __restrict__ val) {
    __shared__ __bf16 cw[DIM*FD];   // [o][c], 32KB
    __shared__ __bf16 xT[64*FD];    // [pix][c], 16KB
    int tid = threadIdx.x;
    int bn = blockIdx.x / (P_/64);
    int pt = blockIdx.x % (P_/64);
    int pixBase = pt*64;
    const __bf16* xsrc = xfbf + ((size_t)bn*P_ + pixBase)*FD;   // contiguous 64x128 tile
    for (int off = tid*8; off < DIM*FD; off += 256*8) cp16(cwbf + off, cw + off);
    for (int off = tid*8; off < 64*FD;  off += 256*8) cp16(xsrc + off, xT + off);
    cp_wait();
    __syncthreads();
    int wave = tid >> 5, lane = tid & 31, l15 = lane & 15, h = lane >> 4;
    v8f acc[4] = {};
#pragma unroll
    for (int ks = 0; ks < 4; ++ks) {
        v16bf a = frag_a(&cw[(wave*16 + l15)*FD + ks*32], h);
#pragma unroll
        for (int t = 0; t < 4; ++t) {
            v16bf b = frag_b(&xT[(t*16 + l15)*FD + ks*32], h);
            acc[t] = __builtin_amdgcn_wmma_f32_16x16x32_bf16(false, a, false, b,
                                                             (short)0, acc[t], false, false);
        }
    }
#pragma unroll
    for (int t = 0; t < 4; ++t)
#pragma unroll
        for (int r = 0; r < 8; ++r) {
            int row = wave*16 + r + 8*h;
            int pix = pixBase + t*16 + l15;
            val[(size_t)bn*DIM*P_ + (size_t)row*P_ + pix] = acc[t][r];
        }
}

// ---------------- K3: homography attention + LN1 (one wave per (b,p)) ----------------
__global__ __launch_bounds__(256) void attn_kernel(const float* __restrict__ Hm,
                                                   const float* __restrict__ val,
                                                   const float* __restrict__ ln1g,
                                                   const float* __restrict__ ln1b,
                                                   float* __restrict__ z1f,
                                                   __bf16* __restrict__ z1bf) {
    int wid  = (blockIdx.x * blockDim.x + threadIdx.x) >> 5;
    int lane = threadIdx.x & 31;
    if (wid >= BP_) return;
    int bb = wid / P_, p = wid % P_;
    int px = p % W_, py = p / W_;
    float X = (float)px / (float)(W_-1) * IMG_W_;
    float Y = (float)py / (float)(H_-1) * IMG_H_;
    int bn0 = bb * N_;

    float qd[4]; float qsq = 0.f;
#pragma unroll
    for (int k = 0; k < 4; ++k) {
        int d = lane + 32*k;
        qd[k] = val[(size_t)bn0*DIM*P_ + (size_t)d*P_ + p];
        qsq += qd[k]*qd[k];
    }
#pragma unroll
    for (int m = 1; m < 32; m <<= 1) qsq += __shfl_xor(qsq, m, 32);
    float qinv = 1.0f / fmaxf(sqrtf(qsq), 1e-12f);

    float vv[N_][4], dotv[N_];
#pragma unroll 1
    for (int n = 0; n < N_; ++n) {
        const float* Hp = Hm + (bn0 + n)*9;
        float hx = Hp[0]*X + Hp[1]*Y + Hp[2];
        float hy = Hp[3]*X + Hp[4]*Y + Hp[5];
        float hz = Hp[6]*X + Hp[7]*Y + Hp[8];
        float u  = hx/hz * ((float)W_ / IMG_W_);
        float vc = hy/hz * ((float)H_ / IMG_H_);
        float x0 = floorf(u), y0 = floorf(vc);
        float wx = u - x0, wy = vc - y0;
        float validf = ((u >= 0.f) && (u <= (float)(W_-1)) &&
                        (vc >= 0.f) && (vc <= (float)(H_-1))) ? 1.0f : 0.0f;
        int x0i = min(max((int)x0, 0), W_-1); int x1i = min(x0i+1, W_-1);
        int y0i = min(max((int)y0, 0), H_-1); int y1i = min(y0i+1, H_-1);
        float w00 = (1.f-wx)*(1.f-wy), w10 = wx*(1.f-wy);
        float w01 = (1.f-wx)*wy,       w11 = wx*wy;
        int i00 = y0i*W_+x0i, i10 = y0i*W_+x1i, i01 = y1i*W_+x0i, i11 = y1i*W_+x1i;
        size_t base = (size_t)(bn0+n)*DIM*P_;
        float ksq = 0.f, dt = 0.f;
#pragma unroll
        for (int k = 0; k < 4; ++k) {
            int d = lane + 32*k;
            const float* vd = val + base + (size_t)d*P_;
            float s = vd[i00]*w00 + vd[i10]*w10 + vd[i01]*w01 + vd[i11]*w11;
            vv[n][k] = s;
            ksq += s*s;
            dt  += s*qd[k];
        }
#pragma unroll
        for (int m = 1; m < 32; m <<= 1) { ksq += __shfl_xor(ksq, m, 32); dt += __shfl_xor(dt, m, 32); }
        float kinv = 1.0f / fmaxf(sqrtf(ksq), 1e-12f);
        dotv[n] = dt * qinv * kinv * validf;
    }
    float mx = dotv[0];
#pragma unroll
    for (int n = 1; n < N_; ++n) mx = fmaxf(mx, dotv[n]);
    float ex[N_]; float se = 0.f;
#pragma unroll
    for (int n = 0; n < N_; ++n) { ex[n] = expf(dotv[n] - mx); se += ex[n]; }
    float inv = 1.0f / se;
    float z[4];
#pragma unroll
    for (int k = 0; k < 4; ++k) z[k] = qd[k];
#pragma unroll
    for (int n = 0; n < N_; ++n)
#pragma unroll
        for (int k = 0; k < 4; ++k) z[k] += ex[n]*inv*vv[n][k];
    float s = 0.f, s2 = 0.f;
#pragma unroll
    for (int k = 0; k < 4; ++k) { s += z[k]; s2 += z[k]*z[k]; }
#pragma unroll
    for (int m = 1; m < 32; m <<= 1) { s += __shfl_xor(s, m, 32); s2 += __shfl_xor(s2, m, 32); }
    float mean = s * (1.0f/DIM);
    float var  = s2 * (1.0f/DIM) - mean*mean;
    float rstd = rsqrtf(var + 1e-5f);
#pragma unroll
    for (int k = 0; k < 4; ++k) {
        int d = lane + 32*k;
        float y = (z[k] - mean)*rstd*ln1g[d] + ln1b[d];
        size_t o = (size_t)wid*DIM + d;
        z1f[o]  = y;
        z1bf[o] = (__bf16)y;
    }
}

// ---------------- K4: MLP GEMM1 + bias + GELU (WMMA bf16) ----------------
__global__ __launch_bounds__(256) void mlp1_kernel(const __bf16* __restrict__ z1bf,
                                                   const __bf16* __restrict__ w1T,
                                                   const float* __restrict__ b1,
                                                   __bf16* __restrict__ hdnbf) {
    __shared__ __bf16 aT[64*DIM];    // [pix][k] 16KB
    __shared__ __bf16 wT[128*DIM];   // [n][k]   32KB
    const int MB = BP_/64;
    int tid = threadIdx.x;
    int nt = blockIdx.x / MB;
    int mb = blockIdx.x % MB;
    int pixBase = mb*64, nBase = nt*128;
    const __bf16* asrc = z1bf + (size_t)pixBase*DIM;     // contiguous 64x128
    const __bf16* wsrc = w1T + (size_t)nBase*DIM;        // contiguous 128x128
    for (int off = tid*8; off < 64*DIM;  off += 256*8) cp16(asrc + off, aT + off);
    for (int off = tid*8; off < 128*DIM; off += 256*8) cp16(wsrc + off, wT + off);
    cp_wait();
    __syncthreads();
    int wave = tid >> 5, lane = tid & 31, l15 = lane & 15, h = lane >> 4;
    int mt = wave & 3, nb = (wave >> 2)*64;
    v8f acc[4] = {};
#pragma unroll
    for (int ks = 0; ks < 4; ++ks) {
        v16bf a = frag_a(&aT[(mt*16 + l15)*DIM + ks*32], h);
#pragma unroll
        for (int t = 0; t < 4; ++t) {
            v16bf b = frag_b(&wT[(nb + t*16 + l15)*DIM + ks*32], h);
            acc[t] = __builtin_amdgcn_wmma_f32_16x16x32_bf16(false, a, false, b,
                                                             (short)0, acc[t], false, false);
        }
    }
#pragma unroll
    for (int t = 0; t < 4; ++t)
#pragma unroll
        for (int r = 0; r < 8; ++r) {
            int nn  = nBase + nb + t*16 + l15;
            int pix = pixBase + mt*16 + r + 8*h;
            float x = acc[t][r] + b1[nn];
            float gl = 0.5f*x*(1.0f + erff(x*0.70710678118f));
            hdnbf[(size_t)pix*(2*DIM) + nn] = (__bf16)gl;
        }
}

// ---------------- K5: MLP GEMM2 + bias + residual + LN2 + transposed store ----------------
__global__ __launch_bounds__(256) void mlp2_kernel(const __bf16* __restrict__ hdnbf,
                                                   const __bf16* __restrict__ w2T,
                                                   const float* __restrict__ b2,
                                                   const float* __restrict__ z1f,
                                                   const float* __restrict__ ln2g,
                                                   const float* __restrict__ ln2b,
                                                   float* __restrict__ out) {
    __shared__ __bf16 aT[64*128];    // 16KB
    __shared__ __bf16 wT[128*128];   // 32KB
    __shared__ float lsum[64], lsum2[64];
    int tid = threadIdx.x;
    int pixBase = blockIdx.x*64;
    if (tid < 64) { lsum[tid] = 0.f; lsum2[tid] = 0.f; }
    int wave = tid >> 5, lane = tid & 31, l15 = lane & 15, h = lane >> 4;
    int mt = wave & 3, nb = (wave >> 2)*64;
    v8f acc[4] = {};
    for (int kc = 0; kc < 2; ++kc) {
        __syncthreads();
        // aT rows: 128 contiguous bf16 from each 256-wide hdn row (16 segs of 8)
        for (int i = tid; i < 64*16; i += 256) {
            int pl = i >> 4, seg = i & 15;
            cp16(hdnbf + (size_t)(pixBase+pl)*(2*DIM) + kc*128 + seg*8,
                 aT + pl*128 + seg*8);
        }
        // wT rows: 128 contiguous bf16 from each 256-wide w2T row
        for (int i = tid; i < 128*16; i += 256) {
            int nl = i >> 4, seg = i & 15;
            cp16(w2T + (size_t)nl*(2*DIM) + kc*128 + seg*8,
                 wT + nl*128 + seg*8);
        }
        cp_wait();
        __syncthreads();
#pragma unroll
        for (int ks = 0; ks < 4; ++ks) {
            v16bf a = frag_a(&aT[(mt*16 + l15)*128 + ks*32], h);
#pragma unroll
            for (int t = 0; t < 4; ++t) {
                v16bf b = frag_b(&wT[(nb + t*16 + l15)*128 + ks*32], h);
                acc[t] = __builtin_amdgcn_wmma_f32_16x16x32_bf16(false, a, false, b,
                                                                 (short)0, acc[t], false, false);
            }
        }
    }
    // bias + residual
#pragma unroll
    for (int t = 0; t < 4; ++t)
#pragma unroll
        for (int r = 0; r < 8; ++r) {
            int d   = nb + t*16 + l15;
            int pix = pixBase + mt*16 + r + 8*h;
            acc[t][r] = acc[t][r] + b2[d] + z1f[(size_t)pix*DIM + d];
        }
    // per-pixel partial LN sums (64 dims per wave), combine across waves in LDS
#pragma unroll
    for (int r = 0; r < 8; ++r) {
        float s = 0.f, s2 = 0.f;
#pragma unroll
        for (int t = 0; t < 4; ++t) { float v = acc[t][r]; s += v; s2 += v*v; }
#pragma unroll
        for (int m = 1; m < 16; m <<= 1) { s += __shfl_xor(s, m, 32); s2 += __shfl_xor(s2, m, 32); }
        if (l15 == 0) {
            int pl = mt*16 + r + 8*h;
            atomicAdd(&lsum[pl], s);
            atomicAdd(&lsum2[pl], s2);
        }
    }
    __syncthreads();
#pragma unroll
    for (int t = 0; t < 4; ++t)
#pragma unroll
        for (int r = 0; r < 8; ++r) {
            int d  = nb + t*16 + l15;
            int pl = mt*16 + r + 8*h;
            int pix = pixBase + pl;
            float mean = lsum[pl] * (1.0f/DIM);
            float var  = lsum2[pl] * (1.0f/DIM) - mean*mean;
            float y = (acc[t][r] - mean)*rsqrtf(var + 1e-5f)*ln2g[d] + ln2b[d];
            int bb = pix / P_, pp = pix % P_;
            out[(size_t)bb*DIM*P_ + (size_t)d*P_ + pp] = y;
        }
}

// ---------------- host launcher ----------------
extern "C" void kernel_launch(void* const* d_in, const int* in_sizes, int n_in,
                              void* d_out, int out_size, void* d_ws, size_t ws_size,
                              hipStream_t stream) {
    const float* feature   = (const float*)d_in[0];
    const float* I_src     = (const float*)d_in[1];
    const float* I_tar_inv = (const float*)d_in[2];
    const float* E         = (const float*)d_in[3];
    const float* dis       = (const float*)d_in[4];
    const float* nrm       = (const float*)d_in[5];
    const float* conv_w    = (const float*)d_in[6];
    const float* bn_gamma  = (const float*)d_in[7];
    const float* bn_beta   = (const float*)d_in[8];
    const float* bn_mean   = (const float*)d_in[9];
    const float* bn_var    = (const float*)d_in[10];
    const float* ln1_g     = (const float*)d_in[11];
    const float* ln1_b     = (const float*)d_in[12];
    const float* ln2_g     = (const float*)d_in[13];
    const float* ln2_b     = (const float*)d_in[14];
    const float* mlp_w1    = (const float*)d_in[15];
    const float* mlp_b1    = (const float*)d_in[16];
    const float* mlp_w2    = (const float*)d_in[17];
    const float* mlp_b2    = (const float*)d_in[18];
    float* out = (float*)d_out;

    // workspace layout (256B aligned)
    char* ws = (char*)d_ws;
    size_t off = 0;
    auto take = [&](size_t bytes) { size_t o = off; off += (bytes + 255) & ~(size_t)255; return o; };
    float*  Hm    = (float*) (ws + take((size_t)B_*N_*9*sizeof(float)));
    __bf16* cwbf  = (__bf16*)(ws + take((size_t)DIM*FD*sizeof(__bf16)));
    __bf16* w1T   = (__bf16*)(ws + take((size_t)DIM*2*DIM*sizeof(__bf16)));
    __bf16* w2T   = (__bf16*)(ws + take((size_t)2*DIM*DIM*sizeof(__bf16)));
    __bf16* xfbf  = (__bf16*)(ws + take((size_t)B_*N_*FD*P_*sizeof(__bf16)));
    float*  val   = (float*) (ws + take((size_t)B_*N_*DIM*P_*sizeof(float)));
    float*  z1f   = (float*) (ws + take((size_t)BP_*DIM*sizeof(float)));
    __bf16* z1bf  = (__bf16*)(ws + take((size_t)BP_*DIM*sizeof(__bf16)));
    __bf16* hdnbf = (__bf16*)(ws + take((size_t)BP_*2*DIM*sizeof(__bf16)));
    (void)ws_size; (void)n_in; (void)in_sizes; (void)out_size;

    setup_kernel<<<1, 256, 0, stream>>>(I_src, I_tar_inv, E, dis, nrm,
                                        conv_w, mlp_w1, mlp_w2,
                                        Hm, cwbf, w1T, w2T);

    int total = B_*N_*FD*P_;
    bnrelu_kernel<<<8192, 256, 0, stream>>>(feature, bn_gamma, bn_beta, bn_mean, bn_var,
                                            xfbf, total);

    conv_kernel<<<B_*N_*(P_/64), 256, 0, stream>>>(xfbf, cwbf, val);

    attn_kernel<<<(BP_*32 + 255)/256, 256, 0, stream>>>(Hm, val, ln1_g, ln1_b, z1f, z1bf);

    mlp1_kernel<<<(BP_/64)*2, 256, 0, stream>>>(z1bf, w1T, mlp_b1, hdnbf);

    mlp2_kernel<<<BP_/64, 256, 0, stream>>>(hdnbf, w2T, mlp_b2, z1f, ln2_g, ln2_b, out);
}
